// SelfAttentionBlock_11029476016302
// MI455X (gfx1250) — compile-verified, matrix operand order checked by reading, not run on hardware
//
#include <hip/hip_runtime.h>

// ---------------------------------------------------------------------------
// Self-attention block for (B=16, C=64, H=64, W=64), N = 4096.
// ~71 GF total, 97% in the attention GEMMs -> bf16 WMMA flash-attention.
// Attention kernel: 4 waves/WG, 32 query rows per wave (2 x 16-row tiles
// sharing every K/V B-fragment -> 2x arithmetic intensity), 128 rows per WG.
// Workspace layout (bytes):
//   [0)    qkvb : bf16 (B,192,4096)  = 24 MB  (raw conv layout)
//   [24MB) KT   : bf16 (B,4096,64)   =  8 MB  (K^T, rows = key idx)
//   [32MB) VT   : bf16 (B,64,4096)   =  8 MB  (V^T, rows = channel)
//   [40MB) Obuf : f32  (B,4096,64)   = 16 MB  (attention output)
// ---------------------------------------------------------------------------

typedef __attribute__((ext_vector_type(16))) __bf16       v16bf;
typedef __attribute__((ext_vector_type(8)))  float        v8f;
typedef __attribute__((ext_vector_type(4)))  unsigned int u32x4;

#define NB   4096
#define CB   64

static __device__ __forceinline__ unsigned short f2bf(float f) {
    unsigned int u = __float_as_uint(f);
    u = (u + 0x7FFFu + ((u >> 16) & 1u)) >> 16;   // round-to-nearest-even
    return (unsigned short)u;
}

union FragU { u32x4 q[2]; v16bf v; };

// Load a 16x32 bf16 A/B fragment. `p` already points at
//   &buf[row*ld + half*8]   (half = lane>>4), 16B aligned.
// Chunks at p and p+16 cover contraction indices {half*8..+7, 16+half*8..+7},
// matching the CDNA5 16-bit A-matrix VGPR layout (ISA 7.12.2).
static __device__ __forceinline__ v16bf load_frag(const unsigned short* p) {
    FragU f;
    f.q[0] = *(const u32x4*)(p);
    f.q[1] = *(const u32x4*)(p + 16);
    return f.v;
}

static __device__ __forceinline__ v8f wmma_bf16(v16bf a, v16bf b, v8f c) {
    return __builtin_amdgcn_wmma_f32_16x16x32_bf16(
        false, a, false, b, (short)0, c, false, false);
}

// ---------------------------------------------------------------------------
// Kernel 1: 1x1-conv QKV projection, fp32 FMA, writes bf16 qkv in conv layout.
// ---------------------------------------------------------------------------
__global__ __launch_bounds__(256) void qkv_kernel(
    const float* __restrict__ x, const float* __restrict__ w,
    const float* __restrict__ bias, unsigned short* __restrict__ qkvb)
{
    int b = blockIdx.x >> 4;                  // 16 n-blocks per batch
    int n = ((blockIdx.x & 15) << 8) + threadIdx.x;
    const float* xb = x + (size_t)b * CB * NB;
    float xr[CB];
#pragma unroll
    for (int c = 0; c < CB; ++c) xr[c] = xb[(size_t)c * NB + n];
    unsigned short* ob = qkvb + (size_t)b * 3 * CB * NB;
    for (int o = 0; o < 3 * CB; ++o) {
        float acc = bias[o];
#pragma unroll
        for (int c = 0; c < CB; ++c) acc = fmaf(w[o * CB + c], xr[c], acc);
        ob[(size_t)o * NB + n] = f2bf(acc);
    }
}

// ---------------------------------------------------------------------------
// Kernel 2a: KT[m][c] = k_part[c][m]   (transpose (64,4096) -> (4096,64))
// ---------------------------------------------------------------------------
__global__ __launch_bounds__(256) void ktrans_kernel(
    const unsigned short* __restrict__ qkvb, unsigned short* __restrict__ KT)
{
    __shared__ unsigned short t[64][65];
    int b  = blockIdx.y;
    int mB = blockIdx.x << 6;
    const unsigned short* src = qkvb + (size_t)b * 3 * CB * NB + (size_t)CB * NB;
    unsigned short* dst = KT + (size_t)b * NB * CB;
    int tx = threadIdx.x & 63, g = threadIdx.x >> 6;
#pragma unroll
    for (int i = 0; i < 16; ++i) {
        int c = (g << 4) + i;
        t[c][tx] = src[(size_t)c * NB + mB + tx];
    }
    __syncthreads();
#pragma unroll
    for (int i = 0; i < 16; ++i) {
        int m = (g << 4) + i;
        dst[(size_t)(mB + m) * CB + tx] = t[tx][m];
    }
}

// ---------------------------------------------------------------------------
// Kernel 2b: VT[c][m] = v_part_flat[m*64 + c]  ((4096,64) -> (64,4096))
// ---------------------------------------------------------------------------
__global__ __launch_bounds__(256) void vtrans_kernel(
    const unsigned short* __restrict__ qkvb, unsigned short* __restrict__ VT)
{
    __shared__ unsigned short t[64][65];
    int b  = blockIdx.y;
    int mB = blockIdx.x << 6;
    const unsigned short* src = qkvb + (size_t)b * 3 * CB * NB + (size_t)2 * CB * NB;
    unsigned short* dst = VT + (size_t)b * CB * NB;
    int tx = threadIdx.x & 63, g = threadIdx.x >> 6;
#pragma unroll
    for (int i = 0; i < 16; ++i) {
        int r = (g << 4) + i;
        t[r][tx] = src[(size_t)(mB + r) * CB + tx];
    }
    __syncthreads();
#pragma unroll
    for (int i = 0; i < 16; ++i) {
        int c = (g << 4) + i;
        dst[(size_t)c * NB + mB + tx] = t[tx][c];
    }
}

// ---------------------------------------------------------------------------
// Kernel 3: flash attention. 4 waves/WG, 32 query rows per wave (2 tiles of
// 16 rows sharing each K/V B-fragment), 128 rows per WG. Online softmax in
// exp2 domain; P transposed D-layout -> A-layout via bank-padded LDS.
// ---------------------------------------------------------------------------
__global__ __launch_bounds__(128) void attn_kernel(
    const unsigned short* __restrict__ qkvb,
    const unsigned short* __restrict__ KT,
    const unsigned short* __restrict__ VT,
    float* __restrict__ Obuf)
{
    __shared__ __align__(16) unsigned short Pt[4 * 32 * 72];  // padded rows

    int tid  = threadIdx.x;
    int wave = tid >> 5, lane = tid & 31;
    int half = lane >> 4, lr = lane & 15;
    int b    = blockIdx.x >> 5;          // 32 row-blocks of 128 per batch
    int tile = blockIdx.x & 31;
    int row0 = (tile << 7) + (wave << 5);   // this wave's first of 32 rows

    const unsigned short* Q  = qkvb + (size_t)b * 3 * CB * NB; // (4096,64) view
    const unsigned short* Kt = KT + (size_t)b * NB * CB;       // (4096,64)
    const unsigned short* Vt = VT + (size_t)b * CB * NB;       // (64,4096)

    // Q fragments: 2 row-tiles x (K=64 -> two 16x32 fragments)
    const unsigned short* qrow = Q + (size_t)(row0 + lr) * CB + half * 8;
    v16bf qa[2][2];
#pragma unroll
    for (int t = 0; t < 2; ++t) {
        const unsigned short* qp = qrow + (size_t)(t << 4) * CB;
        qa[t][0] = load_frag(qp);
        qa[t][1] = load_frag(qp + 32);
    }

    const v8f z = {0,0,0,0,0,0,0,0};
    v8f O[2][4];
    float mst[2][8], lst[2][8];
#pragma unroll
    for (int t = 0; t < 2; ++t) {
#pragma unroll
        for (int jd = 0; jd < 4; ++jd) O[t][jd] = z;
#pragma unroll
        for (int j = 0; j < 8; ++j) { mst[t][j] = -INFINITY; lst[t][j] = 0.0f; }
    }

    const float scale = 0.125f * 1.44269504088896340736f;  // C^-0.5 * log2(e)
    unsigned short* pw = Pt + wave * 32 * 72;

    for (int kt = 0; kt < 64; ++kt) {
        int mbase = kt << 6;

        // ---- S = scale * Q @ K : 2 tiles x 4 accs, B-fragments shared ----
        v8f S[2][4];
        {
            const unsigned short* kb = Kt + (size_t)(mbase + lr) * CB + half * 8;
#pragma unroll
            for (int jn = 0; jn < 4; ++jn) {
                const unsigned short* kp = kb + (size_t)(jn << 4) * CB;
                v16bf b0 = load_frag(kp);
                v16bf b1 = load_frag(kp + 32);
                S[0][jn] = wmma_bf16(qa[0][1], b1, wmma_bf16(qa[0][0], b0, z));
                S[1][jn] = wmma_bf16(qa[1][1], b1, wmma_bf16(qa[1][0], b0, z));
            }
        }

        // ---- online softmax per row-tile ----
#pragma unroll
        for (int t = 0; t < 2; ++t) {
#pragma unroll
            for (int jn = 0; jn < 4; ++jn) S[t][jn] *= scale;
            float mnew[8], alpha[8];
#pragma unroll
            for (int j = 0; j < 8; ++j) {
                float c = fmaxf(fmaxf(S[t][0][j], S[t][1][j]),
                                fmaxf(S[t][2][j], S[t][3][j]));
#pragma unroll
                for (int m = 1; m < 16; m <<= 1)
                    c = fmaxf(c, __shfl_xor(c, m, 32));
                float mn = fmaxf(mst[t][j], c);
                alpha[j] = exp2f(mst[t][j] - mn);
                mnew[j]  = mn;
            }
#pragma unroll
            for (int j = 0; j < 8; ++j) {
                S[t][0][j] = exp2f(S[t][0][j] - mnew[j]);
                S[t][1][j] = exp2f(S[t][1][j] - mnew[j]);
                S[t][2][j] = exp2f(S[t][2][j] - mnew[j]);
                S[t][3][j] = exp2f(S[t][3][j] - mnew[j]);
                float r = (S[t][0][j] + S[t][1][j]) + (S[t][2][j] + S[t][3][j]);
#pragma unroll
                for (int m = 1; m < 16; m <<= 1) r += __shfl_xor(r, m, 32);
                lst[t][j] = lst[t][j] * alpha[j] + r;
                mst[t][j] = mnew[j];
                O[t][0][j] *= alpha[j]; O[t][1][j] *= alpha[j];
                O[t][2][j] *= alpha[j]; O[t][3][j] *= alpha[j];
            }

            // ---- P: D-layout -> bf16 A-layout through LDS (wave-private) ----
#pragma unroll
            for (int j = 0; j < 8; ++j) {
                int r = (t << 4) + (half << 3) + j;   // local row within 32
                pw[r * 72 +  0 + lr] = f2bf(S[t][0][j]);
                pw[r * 72 + 16 + lr] = f2bf(S[t][1][j]);
                pw[r * 72 + 32 + lr] = f2bf(S[t][2][j]);
                pw[r * 72 + 48 + lr] = f2bf(S[t][3][j]);
            }
        }
        asm volatile("" ::: "memory");   // DS ops are in-order per wave
        v16bf pa[2][2];
#pragma unroll
        for (int t = 0; t < 2; ++t) {
            const unsigned short* pr = pw + ((t << 4) + lr) * 72 + half * 8;
            pa[t][0] = load_frag(pr);
            pa[t][1] = load_frag(pr + 32);
        }
        asm volatile("" ::: "memory");

        // ---- O += P @ V : V B-fragments shared across the 2 row-tiles ----
        {
            const unsigned short* vb = Vt + (size_t)lr * NB + mbase + half * 8;
#pragma unroll
            for (int jd = 0; jd < 4; ++jd) {
                const unsigned short* vp = vb + (size_t)(jd << 4) * NB;
                v16bf b0 = load_frag(vp);
                v16bf b1 = load_frag(vp + 32);
                O[0][jd] = wmma_bf16(pa[0][1], b1, wmma_bf16(pa[0][0], b0, O[0][jd]));
                O[1][jd] = wmma_bf16(pa[1][1], b1, wmma_bf16(pa[1][0], b0, O[1][jd]));
            }
        }
    }

    // ---- normalize and store O (f32, (B,4096,64)) ----
    float* Ob = Obuf + (size_t)b * NB * CB;
#pragma unroll
    for (int t = 0; t < 2; ++t) {
        float inv[8];
#pragma unroll
        for (int j = 0; j < 8; ++j) inv[j] = 1.0f / lst[t][j];
#pragma unroll
        for (int jd = 0; jd < 4; ++jd) {
#pragma unroll
            for (int j = 0; j < 8; ++j) {
                int r = row0 + (t << 4) + (half << 3) + j;
                int c = (jd << 4) + lr;
                Ob[(size_t)r * CB + c] = O[t][jd][j] * inv[j];
            }
        }
    }
}

// ---------------------------------------------------------------------------
// Kernel 4: out = w_out @ O^T + b_out + x   (fp32, LDS-tiled, bank-safe)
// ---------------------------------------------------------------------------
__global__ __launch_bounds__(256) void proj_kernel(
    const float* __restrict__ Obuf, const float* __restrict__ x,
    const float* __restrict__ w, const float* __restrict__ bias,
    float* __restrict__ out)
{
    __shared__ float t[64][65];
    int b  = blockIdx.x >> 6;
    int nB = (blockIdx.x & 63) << 6;
    const float* Ob = Obuf + ((size_t)b * NB + nB) * CB;
#pragma unroll
    for (int k = 0; k < 16; ++k) {
        int idx = threadIdx.x + (k << 8);
        t[idx >> 6][idx & 63] = Ob[idx];
    }
    __syncthreads();
    int n = threadIdx.x & 63, og = threadIdx.x >> 6;
#pragma unroll 1
    for (int oi = 0; oi < 16; ++oi) {
        int o = (og << 4) + oi;
        float acc = bias[o];
#pragma unroll
        for (int c = 0; c < CB; ++c) acc = fmaf(w[o * CB + c], t[n][c], acc);
        size_t gi = ((size_t)b * CB + o) * NB + nB + n;
        out[gi] = acc + x[gi];
    }
}

// ---------------------------------------------------------------------------
extern "C" void kernel_launch(void* const* d_in, const int* in_sizes, int n_in,
                              void* d_out, int out_size, void* d_ws, size_t ws_size,
                              hipStream_t stream)
{
    (void)in_sizes; (void)n_in; (void)out_size; (void)ws_size;
    const float* x     = (const float*)d_in[0];
    const float* w_qkv = (const float*)d_in[1];
    const float* b_qkv = (const float*)d_in[2];
    const float* w_out = (const float*)d_in[3];
    const float* b_out = (const float*)d_in[4];
    float* out = (float*)d_out;

    char* ws = (char*)d_ws;
    const size_t B = 16;
    unsigned short* qkvb = (unsigned short*)(ws);                          // 24 MB
    unsigned short* KT   = (unsigned short*)(ws + (((size_t)24) << 20));   //  8 MB
    unsigned short* VT   = (unsigned short*)(ws + (((size_t)32) << 20));   //  8 MB
    float*          Obuf = (float*)        (ws + (((size_t)40) << 20));    // 16 MB

    qkv_kernel<<<dim3(B * 16), dim3(256), 0, stream>>>(x, w_qkv, b_qkv, qkvb);
    ktrans_kernel<<<dim3(64, B), dim3(256), 0, stream>>>(qkvb, KT);
    vtrans_kernel<<<dim3(64, B), dim3(256), 0, stream>>>(qkvb, VT);
    attn_kernel<<<dim3(B * 32), dim3(128), 0, stream>>>(qkvb, KT, VT, Obuf);
    proj_kernel<<<dim3(B * 64), dim3(256), 0, stream>>>(Obuf, x, w_out, b_out, out);
}
